// Attention_82386062672405
// MI455X (gfx1250) — matrix operand order repaired and checked by reading, not compiled, not run on hardware
//
#include <hip/hip_runtime.h>
#include <hip/hip_bf16.h>

typedef __attribute__((ext_vector_type(16))) _Float16 v16h;
typedef __attribute__((ext_vector_type(8)))  _Float16 v8h;
typedef __attribute__((ext_vector_type(4)))  _Float16 v4h;
typedef __attribute__((ext_vector_type(8)))  float    v8f;
typedef __attribute__((ext_vector_type(4)))  unsigned int v4u;
typedef __attribute__((ext_vector_type(8)))  int      v8i;
typedef __attribute__((ext_vector_type(4)))  int      v4i;

#define DEV static __device__ __forceinline__

static constexpr int kB     = 8;
static constexpr int kC     = 192;   // dim
static constexpr int kC3    = 576;   // 3*dim
static constexpr int kHW    = 16384; // 128*128
static constexpr int kW     = 128;
static constexpr int kHeads = 6;
static constexpr int kCH    = 32;    // channels per head
static constexpr int kSeg   = 16;    // split-K segments for the Gram pass

// ---- WMMA fragment helpers (CDNA5 wave32 16x16x32 f16; contiguous-K order, same in A and B) ----

DEV v16h combine16(v8h lo, v8h hi) {
  return __builtin_shufflevector(lo, hi, 0,1,2,3,4,5,6,7,8,9,10,11,12,13,14,15);
}

// A matrix 16x32 f16: lane l holds row m=l&15; 16 contiguous K halves starting
// at (l>>4)*16.  base points at element (m0, k0); rowStride in halves.
DEV v16h loadA(const _Float16* __restrict__ base, int rowStride, int lane) {
  int m  = lane & 15;
  int hi = lane >> 4;
  const _Float16* p = base + (size_t)m * rowStride + hi * 16;
  v8h lo = *(const v8h*)(p);
  v8h hh = *(const v8h*)(p + 8);
  return combine16(lo, hh);
}

// B matrix 32x16 f16: lane l holds column n=l&15; 16 contiguous K halves starting
// at (l>>4)*16.  base points at element (k0, n0); colStride in halves between columns.
DEV v16h loadB(const _Float16* __restrict__ base, int colStride, int lane) {
  int n  = lane & 15;
  int hi = lane >> 4;
  const _Float16* p = base + (size_t)n * colStride + hi * 16;
  v8h lo = *(const v8h*)(p);
  v8h hh = *(const v8h*)(p + 8);
  return combine16(lo, hh);
}

// 16 contiguous K halves from an LDS row (already at the lane's column)
DEV v16h loadB_lds(const _Float16* p) {
  v8h lo = *(const v8h*)(p);
  v8h hh = *(const v8h*)(p + 8);
  return combine16(lo, hh);
}

DEV v8f wmma_f(v16h a, v16h b, v8f c) {
  return __builtin_amdgcn_wmma_f32_16x16x32_f16(false, a, false, b, (short)0, c, false, false);
}

// ---------------- utility kernels ----------------
__global__ void k_cvt_f16(const float* __restrict__ src, _Float16* __restrict__ dst, int n) {
  int i = blockIdx.x * blockDim.x + threadIdx.x;
  if (i < n) dst[i] = (_Float16)src[i];
}

__global__ void k_zero_f32(float* __restrict__ p, int n) {
  int i = blockIdx.x * blockDim.x + threadIdx.x;
  if (i < n) p[i] = 0.f;
}

// ---------------- K0b: x[b][c][n] fp32 -> xt[b][n][c] fp16 (LDS transpose) -------
__global__ void k_xpose(const float* __restrict__ x, _Float16* __restrict__ xt) {
  __shared__ _Float16 tile[kC][65];
  int t  = threadIdx.x;
  int n0 = blockIdx.x * 64;
  int b  = blockIdx.y;
  const float* xb = x + (size_t)b * kC * kHW;
#pragma unroll
  for (int i = 0; i < 48; ++i) {            // 192*64 / 256
    int idx = t + i * 256;
    int c = idx >> 6, j = idx & 63;
    tile[c][j] = (_Float16)xb[(size_t)c * kHW + n0 + j];
  }
  __syncthreads();
  _Float16* xo = xt + ((size_t)b * kHW + n0) * kC;
#pragma unroll
  for (int i = 0; i < 48; ++i) {
    int idx = t + i * 256;
    int j = idx / kC, c = idx % kC;
    xo[(size_t)j * kC + c] = tile[c][j];    // contiguous in c -> coalesced
  }
}

// ---------------- K1: qkv = W(576x192) @ x^T + b  (WMMA f16) ----------------
__global__ void k_qkv_gemm(const _Float16* __restrict__ wh, const _Float16* __restrict__ xt,
                           const float* __restrict__ bias, _Float16* __restrict__ qkvh) {
  int lane = threadIdx.x & 31;
  int wv   = threadIdx.x >> 5;
  int n0 = (blockIdx.x * 8 + wv) * 16;
  int m0 = blockIdx.y * 16;
  int b  = blockIdx.z;
  const _Float16* xb = xt + ((size_t)b * kHW + n0) * kC;
  v8f acc = {};
#pragma unroll
  for (int k0 = 0; k0 < kC; k0 += 32) {
    if (k0 + 32 < kC) __builtin_prefetch(xb + k0 + 32, 0, 3);
    v16h a  = loadA(wh + (size_t)m0 * kC + k0, kC, lane);
    v16h bf = loadB(xb + k0, kC, lane);
    acc = wmma_f(a, bf, acc);
  }
  int n  = n0 + (lane & 15);
  int hi = lane >> 4;
  _Float16* op = qkvh + (size_t)b * kC3 * kHW;
#pragma unroll
  for (int r = 0; r < 8; ++r) {
    int m = m0 + r + 8 * hi;
    op[(size_t)m * kHW + n] = (_Float16)(acc[r] + bias[m]);
  }
}

// ---------------- K2: depthwise 3x3 (SAME) + bias ----------------
__global__ void k_dwconv(const _Float16* __restrict__ qin, const float* __restrict__ w9,
                         const float* __restrict__ bias, _Float16* __restrict__ qout) {
  int n = blockIdx.x * blockDim.x + threadIdx.x;  // 0..16383
  int o = blockIdx.y;
  int b = blockIdx.z;
  int y = n >> 7, xw = n & 127;
  const _Float16* src = qin + ((size_t)b * kC3 + o) * kHW;
  const float* wk = w9 + o * 9;
  float s = bias[o];
#pragma unroll
  for (int dy = -1; dy <= 1; ++dy) {
    int yy = y + dy;
    if (yy < 0 || yy > 127) continue;
#pragma unroll
    for (int dx = -1; dx <= 1; ++dx) {
      int xx = xw + dx;
      if (xx < 0 || xx > 127) continue;
      s += wk[(dy + 1) * 3 + (dx + 1)] * (float)src[yy * kW + xx];
    }
  }
  qout[((size_t)b * kC3 + o) * kHW + n] = (_Float16)s;
}

// ---------------- K3a: split-K partial Gram + norms (WMMA + global f32 atomics) --
// G[c,d] = sum_n q[c,n]k[d,n]; 16 segments over n for occupancy (768 blocks).
__global__ void k_gram(const _Float16* __restrict__ qkvd,
                       float* __restrict__ gramf, float* __restrict__ normf) {
  int lane = threadIdx.x & 31, wv = threadIdx.x >> 5;  // 4 waves: one quadrant each
  int seg = blockIdx.x, bh = blockIdx.y;
  int b = bh / kHeads, h = bh % kHeads;
  int ci = wv >> 1, dj = wv & 1;
  const _Float16* qb = qkvd + ((size_t)(b * kC3 + h * kCH + ci * 16)) * kHW;
  const _Float16* kb = qkvd + ((size_t)(b * kC3 + kC + h * kCH + dj * 16)) * kHW;
  v8f acc = {};
  float psq = 0.f, psk = 0.f;
  int nbeg = seg * (kHW / kSeg);
  for (int n0 = nbeg; n0 < nbeg + kHW / kSeg; n0 += 32) {
    v16h a  = loadA(qb + n0, kHW, lane);   // q rows, K = spatial
    v16h bf = loadB(kb + n0, kHW, lane);   // k rows as B columns
    acc = wmma_f(a, bf, acc);
    if (dj == 0) {                          // each (m,n) chunk seen exactly once
#pragma unroll
      for (int i = 0; i < 16; ++i) { float v = (float)a[i];  psq += v * v; }
    }
    if (ci == 0) {
#pragma unroll
      for (int i = 0; i < 16; ++i) { float v = (float)bf[i]; psk += v * v; }
    }
  }
  if (dj == 0) atomicAdd(&normf[bh * 64 + ci * 16 + (lane & 15)], psq);
  if (ci == 0) atomicAdd(&normf[bh * 64 + 32 + dj * 16 + (lane & 15)], psk);
  int d  = dj * 16 + (lane & 15);
  int hi = lane >> 4;
  float* gp = gramf + (size_t)bh * kCH * kCH;
#pragma unroll
  for (int r = 0; r < 8; ++r) {
    int c = ci * 16 + r + 8 * hi;
    atomicAdd(&gp[c * kCH + d], acc[r]);   // global_atomic_add_f32, L2-resident
  }
}

// ---------------- K3b: attn = sigmoid(T * G / (|q_c||k_d|)) -> f16 ----------------
__global__ void k_sigmoid(const float* __restrict__ gramf, const float* __restrict__ normf,
                          const float* __restrict__ temp, _Float16* __restrict__ attn) {
  int t = threadIdx.x, bh = blockIdx.x;
  int h = bh % kHeads;
  float tp = temp[h];
  const float* gp = gramf + (size_t)bh * kCH * kCH;
  const float* nf = normf + bh * 64;
  _Float16* ap = attn + (size_t)bh * kCH * kCH;
#pragma unroll
  for (int i = 0; i < 4; ++i) {
    int idx = t + i * 256;
    int c = idx >> 5, d = idx & 31;
    float rq = 1.f / fmaxf(sqrtf(nf[c]), 1e-12f);
    float rk = 1.f / fmaxf(sqrtf(nf[32 + d]), 1e-12f);
    float g = gp[idx] * rq * rk * tp;
    ap[idx] = (_Float16)(1.f / (1.f + __expf(-g)));
  }
}

// ---------------- K4: out = P @ (blockdiag(attn) @ v) + pb  (TDM + WMMA) --------
// LDS map: [0..27648)   vS[192][72]  raw v tile (TDM target, 144B padded rows)
//          reused as    yT[64][200]  after vS is dead
//          [27648..53248) vT[64][200] transposed v tile
__global__ void k_out(const _Float16* __restrict__ qkvd, const _Float16* __restrict__ attn,
                      const _Float16* __restrict__ ph, const float* __restrict__ pb,
                      float* __restrict__ out) {
  __shared__ __align__(16) char smem[53248];
  _Float16* vS = (_Float16*)smem;             // [192][72]
  _Float16* vT = (_Float16*)(smem + 27648);   // [64][200]
  _Float16* yT = (_Float16*)smem;             // [64][200], aliases vS
  int t = threadIdx.x, lane = t & 31, wv = t >> 5;
  int n0 = blockIdx.x * 64;
  int b  = blockIdx.y;
  const _Float16* vsrc = qkvd + ((size_t)(b * kC3 + 2 * kC)) * kHW + n0;

  if (wv == 0) {
    // Tensor DMA: 2D tile, 192 rows x 64 halves, tensor row stride 16384 halves,
    // LDS pad 16B after every 128B row (pad_interval=4 -> 32 DWORDs, pad_amount=3 -> 4 DWORDs)
    unsigned lva = (unsigned)(size_t)(void*)vS;               // low 32 bits = LDS offset
    unsigned long long ga = (unsigned long long)(size_t)(const void*)vsrc;
    v4u g0 = {};
    g0[0] = 1u;                                               // count=1 (valid D#)
    g0[1] = lva;                                              // lds_addr
    g0[2] = (unsigned)ga;                                     // global_addr[31:0]
    g0[3] = (unsigned)((ga >> 32) & 0x01FFFFFFull) | (2u << 30); // addr[56:32] | type=2
    v8i g1 = {};
    g1[0] = (int)((1u << 16) | (1u << 20) | (4u << 22) | (3u << 25)); // data_size=2B, pad
    g1[1] = (int)(16384u << 16);                              // tensor_dim0[15:0] @ bits63:48
    g1[2] = (int)((16384u >> 16) | (192u << 16));             // dim0 hi | tensor_dim1 lo
    g1[3] = (int)(64u << 16);                                 // tile_dim0=64 @ bits127:112
    g1[4] = (int)192u;                                        // tile_dim1=192
    g1[5] = (int)16384u;                                      // tensor_dim0_stride lo32
    v4i gz = {};
    v8i gz8 = {};
    __builtin_amdgcn_tensor_load_to_lds(g0, g1, gz, gz, gz8, 0);
    __builtin_amdgcn_s_wait_tensorcnt(0);
  }
  __syncthreads();
  // in-LDS transpose vS[c][j] -> vT[j][c]
#pragma unroll
  for (int i = 0; i < 48; ++i) {
    int idx = t + i * 256;
    int c = idx % kC, j = idx / kC;
    vT[j * 200 + c] = vS[c * 72 + j];
  }
  __syncthreads();
  // stage 1: y_h = attn_h(32x32) @ v_h(32x64); K=32 == one WMMA per 16x16 tile
  const _Float16* ab = attn + (size_t)(b * kHeads) * kCH * kCH;
#pragma unroll
  for (int i = 0; i < 6; ++i) {             // 48 tiles over 8 waves
    int tile = wv * 6 + i;
    int head = tile >> 3, rest = tile & 7, mi = rest >> 2, nj = rest & 3;
    v16h a = loadA(ab + (size_t)head * kCH * kCH + mi * 16 * kCH, kCH, lane);
    int j  = nj * 16 + (lane & 15);
    int hi = lane >> 4;
    v16h bf = loadB_lds(&vT[j * 200 + head * kCH + hi * 16]);
    v8f acc = {};
    acc = wmma_f(a, bf, acc);
#pragma unroll
    for (int r = 0; r < 8; ++r)
      yT[j * 200 + head * kCH + mi * 16 + r + 8 * hi] = (_Float16)acc[r];
  }
  __syncthreads();
  // stage 2: out = P(192x192) @ y(192x64) + pb, K=192 in 6 WMMA steps
#pragma unroll
  for (int i = 0; i < 6; ++i) {
    int tile = wv * 6 + i;
    int oi = tile >> 2, nj = tile & 3;
    int j  = nj * 16 + (lane & 15);
    int hi = lane >> 4;
    v8f acc = {};
#pragma unroll
    for (int kc = 0; kc < kC; kc += 32) {
      v16h a  = loadA(ph + (size_t)oi * 16 * kC + kc, kC, lane);
      v16h bf = loadB_lds(&yT[j * 200 + kc + hi * 16]);
      acc = wmma_f(a, bf, acc);
    }
#pragma unroll
    for (int r = 0; r < 8; ++r) {
      int o = oi * 16 + r + 8 * hi;
      out[((size_t)b * kC + o) * kHW + n0 + j] = acc[r] + pb[o];
    }
  }
}

extern "C" void kernel_launch(void* const* d_in, const int* in_sizes, int n_in,
                              void* d_out, int out_size, void* d_ws, size_t ws_size,
                              hipStream_t stream) {
  (void)in_sizes; (void)n_in; (void)out_size; (void)ws_size;
  const float* x      = (const float*)d_in[0];
  const float* temp   = (const float*)d_in[1];
  const float* qkv_w  = (const float*)d_in[2];
  const float* qkv_b  = (const float*)d_in[3];
  const float* dw_w   = (const float*)d_in[4];
  const float* dw_b   = (const float*)d_in[5];
  const float* proj_w = (const float*)d_in[6];
  const float* proj_b = (const float*)d_in[7];
  float* out = (float*)d_out;

  _Float16* ws    = (_Float16*)d_ws;
  _Float16* xt    = ws;                                       // [8][16384][192]
  _Float16* qkvh  = xt   + (size_t)kB * kHW * kC;             // [8][576][16384]
  _Float16* qkvd  = qkvh + (size_t)kB * kC3 * kHW;            // [8][576][16384]
  _Float16* attn  = qkvd + (size_t)kB * kC3 * kHW;            // [48][32][32]
  _Float16* wh    = attn + (size_t)kB * kHeads * kCH * kCH;   // [576][192]
  _Float16* ph    = wh   + (size_t)kC3 * kC;                  // [192][192]
  float*    gramf = (float*)(ph + (size_t)kC * kC);           // [48][32][32] f32
  float*    normf = gramf + (size_t)kB * kHeads * kCH * kCH;  // [48][64] f32

  const int nAccum = kB * kHeads * kCH * kCH + kB * kHeads * 64; // 52224
  k_zero_f32<<<nAccum / 256, 256, 0, stream>>>(gramf, nAccum);
  k_cvt_f16<<<(kC3 * kC) / 256, 256, 0, stream>>>(qkv_w, wh, kC3 * kC);
  k_cvt_f16<<<(kC * kC) / 256, 256, 0, stream>>>(proj_w, ph, kC * kC);
  k_xpose<<<dim3(kHW / 64, kB), 256, 0, stream>>>(x, xt);
  k_qkv_gemm<<<dim3(kHW / 16 / 8, kC3 / 16, kB), 256, 0, stream>>>(wh, xt, qkv_b, qkvh);
  k_dwconv<<<dim3(kHW / 256, kC3, kB), 256, 0, stream>>>(qkvh, dw_w, dw_b, qkvd);
  k_gram<<<dim3(kSeg, kB * kHeads), 128, 0, stream>>>(qkvd, gramf, normf);
  k_sigmoid<<<kB * kHeads, 256, 0, stream>>>(gramf, normf, temp, attn);
  k_out<<<dim3(kHW / 64, kB), 256, 0, stream>>>(qkvd, attn, ph, proj_b, out);
}
